// BitLinear158_77704548319929
// MI455X (gfx1250) — matrix operand order repaired and checked by reading, not compiled, not run on hardware
//
#include <hip/hip_runtime.h>
#include <hip/hip_bf16.h>

// BitLinear 1.58b forward: out = x @ quantize(W)^T + b
// x: (32768, 2048) f32, W: (2048, 2048) f32, b: (2048,) f32, out: (32768, 2048) f32
// gamma = mean(|W|); Wq = clip(round(W/(gamma+eps)), -1, 1)

typedef __attribute__((ext_vector_type(16))) _Float16 v16h;
typedef __attribute__((ext_vector_type(8)))  _Float16 v8h;
typedef __attribute__((ext_vector_type(8)))  float    v8f;
typedef __attribute__((ext_vector_type(4)))  float    v4f;

#define EPS_BL 1e-5f

constexpr int K_DIM = 2048;
constexpr int N_DIM = 2048;
constexpr int M_DIM = 8 * 4096;          // 32768
constexpr int BM = 128, BN = 128, BK = 32;
constexpr int NKT = K_DIM / BK;          // 64 K-iterations (even)
constexpr int LDS_STRIDE = BK + 8;       // pad rows: 40 f16 = 80 B, breaks bank conflicts
constexpr int LDS_TILE = BM * LDS_STRIDE;

// ---------------- gamma = mean(|W|), deterministic two-pass ----------------

__global__ __launch_bounds__(256) void bl_abs_partial(const float* __restrict__ W,
                                                      float* __restrict__ partial, int n) {
    __shared__ float sdata[256];
    float s = 0.0f;
    for (int i = blockIdx.x * blockDim.x + threadIdx.x; i < n; i += gridDim.x * blockDim.x)
        s += fabsf(W[i]);
    sdata[threadIdx.x] = s;
    __syncthreads();
    for (int off = 128; off > 0; off >>= 1) {
        if ((int)threadIdx.x < off) sdata[threadIdx.x] += sdata[threadIdx.x + off];
        __syncthreads();
    }
    if (threadIdx.x == 0) partial[blockIdx.x] = sdata[0];
}

__global__ __launch_bounds__(256) void bl_gamma_final(const float* __restrict__ partial,
                                                      float* __restrict__ gamma, int n_elems) {
    __shared__ float sdata[256];
    sdata[threadIdx.x] = partial[threadIdx.x];
    __syncthreads();
    for (int off = 128; off > 0; off >>= 1) {
        if ((int)threadIdx.x < off) sdata[threadIdx.x] += sdata[threadIdx.x + off];
        __syncthreads();
    }
    if (threadIdx.x == 0) gamma[0] = sdata[0] / (float)n_elems;
}

// ---------------- main WMMA GEMM with fused ternary quantization ----------------

// LDS-only workgroup barrier: waits DScnt (staging stores visible) but leaves
// in-flight global prefetch loads (private VGPR destinations) outstanding,
// unlike __syncthreads() which lowers to s_wait_loadcnt_dscnt 0x0.
__device__ __forceinline__ void wg_barrier_lds() {
    asm volatile("s_wait_dscnt 0x0\n\t"
                 "s_barrier_signal -1\n\t"
                 "s_barrier_wait -1" ::: "memory");
}

__device__ __forceinline__ _Float16 bl_quant(float w, float inv) {
    float q = rintf(w * inv);                 // round-to-nearest-even
    q = fminf(1.0f, fmaxf(-1.0f, q));         // clip to {-1,0,+1}
    return (_Float16)q;
}

__device__ __forceinline__ void bl_prefetch(const float* __restrict__ px,
                                            const float* __restrict__ pw,
                                            v4f (&xr)[4], v4f (&wr)[4]) {
#pragma unroll
    for (int i = 0; i < 4; ++i) {
        xr[i] = *(const v4f*)(px + 4 * i);
        wr[i] = *(const v4f*)(pw + 4 * i);
    }
}

__device__ __forceinline__ void bl_stage(_Float16* __restrict__ ad, _Float16* __restrict__ bd,
                                         const v4f (&xr)[4], const v4f (&wr)[4], float inv) {
    v8h h0, h1, q0, q1;
#pragma unroll
    for (int i = 0; i < 4; ++i) {
        h0[i]     = (_Float16)xr[0][i];
        h0[i + 4] = (_Float16)xr[1][i];
        h1[i]     = (_Float16)xr[2][i];
        h1[i + 4] = (_Float16)xr[3][i];
        q0[i]     = bl_quant(wr[0][i], inv);
        q0[i + 4] = bl_quant(wr[1][i], inv);
        q1[i]     = bl_quant(wr[2][i], inv);
        q1[i + 4] = bl_quant(wr[3][i], inv);
    }
    *(v8h*)(ad)     = h0;
    *(v8h*)(ad + 8) = h1;
    *(v8h*)(bd)     = q0;
    *(v8h*)(bd + 8) = q1;
}

__global__ __launch_bounds__(256)
void bl_gemm_wmma(const float* __restrict__ x, const float* __restrict__ W,
                  const float* __restrict__ bias, const float* __restrict__ gptr,
                  float* __restrict__ out) {
    // double-buffered LDS stages (ping-pong, one barrier per K-iteration)
    __shared__ _Float16 As[2 * LDS_TILE];
    __shared__ _Float16 Bs[2 * LDS_TILE];

    const float inv = 1.0f / (gptr[0] + EPS_BL);

    const int tid  = threadIdx.x;
    const int bm0  = blockIdx.y * BM;
    const int bn0  = blockIdx.x * BN;

    const int wave = tid >> 5;      // 0..7 (wave32)
    const int lane = tid & 31;
    const int wm   = wave & 3;      // 4 waves along M: 32 rows each
    const int wn   = wave >> 2;     // 2 waves along N: 64 cols each

    v8f acc[2][4] = {};

    // cooperative LDS loaders: thread -> (row 0..127, 16-element K half)
    const int lrow  = tid >> 1;
    const int lhalf = (tid & 1) * 16;

    const float* __restrict__ xrow = x + (size_t)(bm0 + lrow) * K_DIM + lhalf;
    const float* __restrict__ wrow = W + (size_t)(bn0 + lrow) * K_DIM + lhalf;
    const int sdst = lrow * LDS_STRIDE + lhalf;

    // fragment addressing (ISA 7.12.2, 16-bit operands, wave32):
    // A: lane L holds row M=L%16; lanes 0-15 K in {0..7,16..23}, lanes 16-31 K in {8..15,24..31}
    // B: lane L holds col N=L%16; lanes 0-15 K in {0..7,16..23}, lanes 16-31 K in {8..15,24..31}
    const int lmod   = lane & 15;
    const int akoff  = (lane >> 4) * 8;   // 0 or 8
    const int bkoff  = (lane >> 4) * 8;   // 0 or 8

    const int aoff0 = (wm * 32 + lmod) * LDS_STRIDE;                 // A rows, mt stride 16*LDS_STRIDE
    const int boff0 = (wn * 64 + lmod) * LDS_STRIDE + bkoff;         // B cols, nt stride 16*LDS_STRIDE

    // two named in-flight register sets -> compile-time ping-pong (no v_movreld)
    v4f x0[4], w0[4], x1[4], w1[4];

    bl_prefetch(xrow, wrow, x0, w0);   // tile 0

    auto compute_stage = [&](int stagebase) {
        v16h afrag[2], bfrag[4];
#pragma unroll
        for (int mt = 0; mt < 2; ++mt) {
            const _Float16* p = &As[stagebase + aoff0 + mt * 16 * LDS_STRIDE];
            v8h lo = *(const v8h*)(p + akoff);
            v8h hi = *(const v8h*)(p + 16 + akoff);
#pragma unroll
            for (int i = 0; i < 8; ++i) { afrag[mt][i] = lo[i]; afrag[mt][i + 8] = hi[i]; }
        }
#pragma unroll
        for (int nt = 0; nt < 4; ++nt) {
            const _Float16* p = &Bs[stagebase + boff0 + nt * 16 * LDS_STRIDE];
            v8h lo = *(const v8h*)(p);
            v8h hi = *(const v8h*)(p + 16);
#pragma unroll
            for (int i = 0; i < 8; ++i) { bfrag[nt][i] = lo[i]; bfrag[nt][i + 8] = hi[i]; }
        }
#pragma unroll
        for (int mt = 0; mt < 2; ++mt)
#pragma unroll
            for (int nt = 0; nt < 4; ++nt)
                acc[mt][nt] = __builtin_amdgcn_wmma_f32_16x16x32_f16(
                    false, afrag[mt], false, bfrag[nt],
                    (short)0, acc[mt][nt], false, false);
    };

    // main loop: 2 K-tiles per trip, all ping-pong indices compile-time constant
    for (int kt = 0; kt < NKT; kt += 2) {
        // ---- sub-iteration 0: stage set0 -> LDS stage 0, prefetch kt+1 -> set1 ----
        bl_prefetch(xrow + (kt + 1) * BK, wrow + (kt + 1) * BK, x1, w1);
        bl_stage(&As[sdst], &Bs[sdst], x0, w0, inv);
        wg_barrier_lds();
        compute_stage(0);

        // ---- sub-iteration 1: stage set1 -> LDS stage 1, prefetch kt+2 -> set0 ----
        if (kt + 2 < NKT)
            bl_prefetch(xrow + (kt + 2) * BK, wrow + (kt + 2) * BK, x0, w0);
        bl_stage(&As[LDS_TILE + sdst], &Bs[LDS_TILE + sdst], x1, w1, inv);
        wg_barrier_lds();
        compute_stage(LDS_TILE);
    }

    // ---- epilogue: bias + store ----
    // C/D layout (ISA 7.12.2): VGPR v: lanes 0-15 -> M=v, lanes 16-31 -> M=v+8; N = lane%16
    const int rbase = (lane >> 4) * 8;
#pragma unroll
    for (int nt = 0; nt < 4; ++nt) {
        const int gn = bn0 + wn * 64 + nt * 16 + lmod;
        const float bv = bias[gn];
#pragma unroll
        for (int mt = 0; mt < 2; ++mt) {
            const int gm0 = bm0 + wm * 32 + mt * 16 + rbase;
            v8f c = acc[mt][nt];
#pragma unroll
            for (int v = 0; v < 8; ++v)
                out[(size_t)(gm0 + v) * N_DIM + gn] = c[v] + bv;
        }
    }
}

// ---------------- launch ----------------

extern "C" void kernel_launch(void* const* d_in, const int* in_sizes, int n_in,
                              void* d_out, int out_size, void* d_ws, size_t ws_size,
                              hipStream_t stream) {
    const float* x = (const float*)d_in[0];
    const float* W = (const float*)d_in[1];
    const float* b = (const float*)d_in[2];
    float* out     = (float*)d_out;

    float* partial = (float*)d_ws;       // 256 floats
    float* gamma   = partial + 256;      // 1 float

    const int KN = K_DIM * N_DIM;        // 4M elements of W

    bl_abs_partial<<<256, 256, 0, stream>>>(W, partial, KN);
    bl_gamma_final<<<1, 256, 0, stream>>>(partial, gamma, KN);

    dim3 grid(N_DIM / BN, M_DIM / BM);   // (16, 256)
    bl_gemm_wmma<<<grid, 256, 0, stream>>>(x, W, b, gamma, out);
}
